// LRU_46935402611365
// MI455X (gfx1250) — matrix operand order repaired and checked by reading, not compiled, not run on hardware
//
#include <hip/hip_runtime.h>

// ---------------------------------------------------------------------------
// LRU on MI455X (gfx1250): 2 fused bf16-WMMA NT GEMMs + chunked parallel scan.
// All intermediates bf16 & L2-resident (192MB); accumulation/recurrence in f32.
// ---------------------------------------------------------------------------

#define L_SEQ   16384
#define D_MOD   1024
#define D_HID   1024
#define NCHUNK  128
#define CHLEN   128          // L_SEQ / NCHUNK

typedef __attribute__((ext_vector_type(16))) __bf16 v16bf;
typedef __attribute__((ext_vector_type(8)))  float  v8f;

union AB16 { v16bf v; uint4 u[2]; };

// ---------------------------------------------------------------------------
// Setup: lambda = exp(-exp(nu) + i exp(theta)), gamma = exp(gamma_log),
// lambda^CHLEN in closed form (needed by the chunk-combine pass).
// ---------------------------------------------------------------------------
__global__ void k_setup(const float* __restrict__ nu_log,
                        const float* __restrict__ th_log,
                        const float* __restrict__ g_log,
                        float* __restrict__ lam_re, float* __restrict__ lam_im,
                        float* __restrict__ gamma,
                        float* __restrict__ lamN_re, float* __restrict__ lamN_im) {
  int h = blockIdx.x * blockDim.x + threadIdx.x;
  if (h >= D_HID) return;
  float nu = expf(nu_log[h]);
  float th = expf(th_log[h]);
  float r  = expf(-nu);
  lam_re[h] = r * cosf(th);
  lam_im[h] = r * sinf(th);
  gamma[h]  = expf(g_log[h]);
  float rN = expf(-(float)CHLEN * nu);
  float aN = (float)CHLEN * th;
  lamN_re[h] = rN * cosf(aN);
  lamN_im[h] = rN * sinf(aN);
}

// x (f32) -> bf16
__global__ void k_pack_x(const float* __restrict__ x, __bf16* __restrict__ xb) {
  size_t n = (size_t)L_SEQ * D_MOD;
  size_t stride = (size_t)gridDim.x * blockDim.x;
  for (size_t i = (size_t)blockIdx.x * blockDim.x + threadIdx.x; i < n; i += stride)
    xb[i] = (__bf16)x[i];
}

// Bcat [2H, D]: rows 0..H-1 = gamma*B_re, rows H..2H-1 = gamma*B_im
__global__ void k_pack_B(const float* __restrict__ B_re, const float* __restrict__ B_im,
                         const float* __restrict__ gamma, __bf16* __restrict__ Bcat) {
  size_t n = (size_t)D_HID * D_MOD;
  size_t stride = (size_t)gridDim.x * blockDim.x;
  for (size_t i = (size_t)blockIdx.x * blockDim.x + threadIdx.x; i < n; i += stride) {
    float g = gamma[i / D_MOD];
    Bcat[i]     = (__bf16)(g * B_re[i]);
    Bcat[n + i] = (__bf16)(g * B_im[i]);
  }
}

// McatT [D, 2H]: McatT[d, 0:H] = C_re[d,:], McatT[d, H:2H] = -C_im[d,:]
// so y = Hcat @ McatT^T gives Re(h @ C^T).
__global__ void k_pack_C(const float* __restrict__ C_re, const float* __restrict__ C_im,
                         __bf16* __restrict__ McatT) {
  size_t n = (size_t)D_MOD * D_HID;
  size_t stride = (size_t)gridDim.x * blockDim.x;
  for (size_t i = (size_t)blockIdx.x * blockDim.x + threadIdx.x; i < n; i += stride) {
    size_t d = i / D_HID;                 // i = d*H + h
    McatT[i + d * D_HID]         = (__bf16)( C_re[i]);   // [d*2H + h]
    McatT[i + d * D_HID + D_HID] = (__bf16)(-C_im[i]);   // [d*2H + H + h]
  }
}

// ---------------------------------------------------------------------------
// NT GEMM: C[M,N] = A[M,K] @ B[N,K]^T, bf16 inputs, f32 accumulate via
// v_wmma_f32_16x16x32_bf16. One wave32 per 64x32 tile (4x2 WMMA tiles).
// A-frag per lane: rows m = lane&15; K-elems {half*8+0..7, 16+half*8+0..7}
// B-frag per lane: cols n = lane&15; K-elems {half*8+0..7, 16+half*8+0..7}
// D-frag per lane: m = half*8 + r (r=0..7), n = lane&15
// ---------------------------------------------------------------------------
template <bool EPI>
__global__ __launch_bounds__(32) void gemm_nt_bf16(
    const __bf16* __restrict__ A, int lda,
    const __bf16* __restrict__ B, int ldb,
    int K, int N,
    __bf16* __restrict__ Cb,            // !EPI: bf16 output
    float*  __restrict__ Cf,            //  EPI: f32 output
    const float* __restrict__ Dp,       //  EPI: y += Dp[n]*X[m,n]
    const float* __restrict__ X) {
  const int lane = threadIdx.x & 31;
  const int half = lane >> 4;
  const int l16  = lane & 15;
  const int mBase = blockIdx.x * 64;
  const int nBase = blockIdx.y * 32;

  v8f acc[4][2] = {};

  for (int k = 0; k < K; k += 32) {
    AB16 a[4], b[2];
#pragma unroll
    for (int i = 0; i < 4; ++i) {
      const __bf16* pa = A + (size_t)(mBase + 16 * i + l16) * lda + (k + half * 8);
      a[i].u[0] = *(const uint4*)(pa);
      a[i].u[1] = *(const uint4*)(pa + 16);
    }
#pragma unroll
    for (int j = 0; j < 2; ++j) {
      const __bf16* pb = B + (size_t)(nBase + 16 * j + l16) * ldb + (k + half * 8);
      b[j].u[0] = *(const uint4*)(pb);
      b[j].u[1] = *(const uint4*)(pb + 16);
    }
    if (k + 32 < K) {
      // gfx1250 global_prefetch_b8; locality=3 -> near-cache (WGP) scope so the
      // next K-slab is pulled through GL2 into the WGP cache (data is L2-resident).
      __builtin_prefetch(A + (size_t)(mBase + l16) * lda + k + 32, 0, 3);
      __builtin_prefetch(B + (size_t)(nBase + l16) * ldb + k + 32, 0, 3);
    }
#pragma unroll
    for (int i = 0; i < 4; ++i)
#pragma unroll
      for (int j = 0; j < 2; ++j)
        acc[i][j] = __builtin_amdgcn_wmma_f32_16x16x32_bf16(
            false, a[i].v, false, b[j].v, (short)0, acc[i][j], false, false);
  }

#pragma unroll
  for (int i = 0; i < 4; ++i)
#pragma unroll
    for (int j = 0; j < 2; ++j)
#pragma unroll
      for (int r = 0; r < 8; ++r) {
        int m = mBase + 16 * i + half * 8 + r;
        int n = nBase + 16 * j + l16;
        size_t idx = (size_t)m * N + n;
        if constexpr (EPI) {
          Cf[idx] = acc[i][j][r] + Dp[n] * X[idx];
        } else {
          Cb[idx] = (__bf16)acc[i][j][r];
        }
      }
}

// ---------------------------------------------------------------------------
// Chunked parallel scan of h_t = lambda*h_{t-1} + Bu_t (complex, diagonal).
// Pass A: per-chunk local scan from zero -> chunk_b.
// Pass B: sequential combine over 128 chunks (lambda^CHLEN closed form)
//         -> per-chunk carry-in + final hidden state (d_out[0:2048] interleaved).
// Pass C: replay chunk with carry-in, overwrite Bu in place with h (bf16).
// ---------------------------------------------------------------------------
__global__ void k_scan_chunks(const __bf16* __restrict__ Bu,
                              const float* __restrict__ lam_re,
                              const float* __restrict__ lam_im,
                              float* __restrict__ chunk_b) {
  int h = blockIdx.x * blockDim.x + threadIdx.x;
  int c = blockIdx.y;
  float lr = lam_re[h], li = lam_im[h];
  float br = 0.f, bi = 0.f;
  const __bf16* p = Bu + (size_t)c * CHLEN * (2 * D_HID);
  for (int t = 0; t < CHLEN; ++t) {
    float ur = (float)p[(size_t)t * (2 * D_HID) + h];
    float ui = (float)p[(size_t)t * (2 * D_HID) + D_HID + h];
    float nr = lr * br - li * bi + ur;
    float ni = lr * bi + li * br + ui;
    br = nr; bi = ni;
  }
  chunk_b[(size_t)c * (2 * D_HID) + h]         = br;
  chunk_b[(size_t)c * (2 * D_HID) + D_HID + h] = bi;
}

__global__ void k_scan_carry(const float* __restrict__ chunk_b,
                             const float* __restrict__ lamN_re,
                             const float* __restrict__ lamN_im,
                             float* __restrict__ chunk_init,
                             float* __restrict__ out_hidden) {
  int h = blockIdx.x * blockDim.x + threadIdx.x;
  if (h >= D_HID) return;
  float Lr = lamN_re[h], Li = lamN_im[h];
  float cr = 0.f, ci = 0.f;
  for (int c = 0; c < NCHUNK; ++c) {
    chunk_init[(size_t)c * (2 * D_HID) + h]         = cr;
    chunk_init[(size_t)c * (2 * D_HID) + D_HID + h] = ci;
    float br = chunk_b[(size_t)c * (2 * D_HID) + h];
    float bi = chunk_b[(size_t)c * (2 * D_HID) + D_HID + h];
    float nr = Lr * cr - Li * ci + br;
    float ni = Lr * ci + Li * cr + bi;
    cr = nr; ci = ni;
  }
  out_hidden[2 * h]     = cr;   // complex64 interleaved
  out_hidden[2 * h + 1] = ci;
}

__global__ void k_scan_apply(__bf16* __restrict__ BuH,
                             const float* __restrict__ lam_re,
                             const float* __restrict__ lam_im,
                             const float* __restrict__ chunk_init) {
  int h = blockIdx.x * blockDim.x + threadIdx.x;
  int c = blockIdx.y;
  float lr = lam_re[h], li = lam_im[h];
  float sr = chunk_init[(size_t)c * (2 * D_HID) + h];
  float si = chunk_init[(size_t)c * (2 * D_HID) + D_HID + h];
  __bf16* p = BuH + (size_t)c * CHLEN * (2 * D_HID);
  for (int t = 0; t < CHLEN; ++t) {
    size_t i0 = (size_t)t * (2 * D_HID) + h;
    size_t i1 = i0 + D_HID;
    float ur = (float)p[i0];
    float ui = (float)p[i1];
    float nr = lr * sr - li * si + ur;
    float ni = lr * si + li * sr + ui;
    sr = nr; si = ni;
    p[i0] = (__bf16)sr;     // in-place: Bu slot becomes h_t
    p[i1] = (__bf16)si;
  }
}

// ---------------------------------------------------------------------------
extern "C" void kernel_launch(void* const* d_in, const int* in_sizes, int n_in,
                              void* d_out, int out_size, void* d_ws, size_t ws_size,
                              hipStream_t stream) {
  (void)in_sizes; (void)n_in; (void)out_size; (void)ws_size;
  const float* x      = (const float*)d_in[0];
  const float* nu_log = (const float*)d_in[1];
  const float* th_log = (const float*)d_in[2];
  const float* g_log  = (const float*)d_in[3];
  const float* B_re   = (const float*)d_in[4];
  const float* B_im   = (const float*)d_in[5];
  const float* C_re   = (const float*)d_in[6];
  const float* C_im   = (const float*)d_in[7];
  const float* Dp     = (const float*)d_in[8];
  float* out = (float*)d_out;

  char* ws = (char*)d_ws;
  size_t off = 0;
  auto take = [&](size_t bytes) -> char* {
    off = (off + 255) & ~(size_t)255;
    char* p = ws + off;
    off += bytes;
    return p;
  };

  float*  lam_re  = (float*)take((size_t)D_HID * 4);
  float*  lam_im  = (float*)take((size_t)D_HID * 4);
  float*  gamma   = (float*)take((size_t)D_HID * 4);
  float*  lamN_re = (float*)take((size_t)D_HID * 4);
  float*  lamN_im = (float*)take((size_t)D_HID * 4);
  __bf16* Xbf     = (__bf16*)take((size_t)L_SEQ * D_MOD * 2);        // 32 MB
  __bf16* Bcat    = (__bf16*)take((size_t)2 * D_HID * D_MOD * 2);    //  4 MB
  __bf16* McatT   = (__bf16*)take((size_t)D_MOD * 2 * D_HID * 2);    //  4 MB
  __bf16* BuH     = (__bf16*)take((size_t)L_SEQ * 2 * D_HID * 2);    // 64 MB (Bu, then h in place)
  float*  chunk_b    = (float*)take((size_t)NCHUNK * 2 * D_HID * 4); //  1 MB
  float*  chunk_init = (float*)take((size_t)NCHUNK * 2 * D_HID * 4); //  1 MB

  k_setup<<<(D_HID + 255) / 256, 256, 0, stream>>>(nu_log, th_log, g_log,
      lam_re, lam_im, gamma, lamN_re, lamN_im);
  k_pack_x<<<2048, 256, 0, stream>>>(x, Xbf);
  k_pack_B<<<2048, 256, 0, stream>>>(B_re, B_im, gamma, Bcat);
  k_pack_C<<<2048, 256, 0, stream>>>(C_re, C_im, McatT);

  // Bu_cat[L, 2H] = Xbf[L, D] @ Bcat[2H, D]^T
  gemm_nt_bf16<false><<<dim3(L_SEQ / 64, (2 * D_HID) / 32), 32, 0, stream>>>(
      Xbf, D_MOD, Bcat, D_MOD, /*K=*/D_MOD, /*N=*/2 * D_HID,
      BuH, nullptr, nullptr, nullptr);

  k_scan_chunks<<<dim3(D_HID / 256, NCHUNK), 256, 0, stream>>>(BuH, lam_re, lam_im, chunk_b);
  k_scan_carry<<<(D_HID + 255) / 256, 256, 0, stream>>>(chunk_b, lamN_re, lamN_im,
                                                        chunk_init, out);
  k_scan_apply<<<dim3(D_HID / 256, NCHUNK), 256, 0, stream>>>(BuH, lam_re, lam_im, chunk_init);

  // y[L, D] = Hcat[L, 2H] @ McatT[D, 2H]^T + Dp*x  -> d_out + 2048
  gemm_nt_bf16<true><<<dim3(L_SEQ / 64, D_MOD / 32), 32, 0, stream>>>(
      BuH, 2 * D_HID, McatT, 2 * D_HID, /*K=*/2 * D_HID, /*N=*/D_MOD,
      nullptr, out + 2 * D_HID, Dp, x);
}
